// BiasedMultiHeadAttention_24335284699523
// MI455X (gfx1250) — matrix operand order repaired and checked by reading, not compiled, not run on hardware
//
#include <hip/hip_runtime.h>
#include <hip/hip_bf16.h>

// ---------------------------------------------------------------------------
// BiasedMultiHeadAttention forward for gfx1250 (MI455X), f16 WMMA + f32 accum
// Async global->LDS double buffering where the toolchain exposes it.
// ---------------------------------------------------------------------------

#define cB 2
#define cL 2048
#define cE 1024
#define cH 16
#define cD 64

typedef __attribute__((ext_vector_type(16))) _Float16 v16h;
typedef __attribute__((ext_vector_type(8)))  float    v8f;
typedef __attribute__((ext_vector_type(4)))  _Float16 h4;

struct alignas(16) U128 { unsigned int w[4]; };
union F16x16 { U128 u[2]; v16h v; };

#if defined(__has_builtin)
#if __has_builtin(__builtin_amdgcn_global_load_async_to_lds_b128) && \
    __has_builtin(__builtin_amdgcn_s_wait_asynccnt)
#define USE_ASYNC 1
#endif
#endif
#ifndef USE_ASYNC
#define USE_ASYNC 0
#endif

#if USE_ASYNC
typedef int v4i __attribute__((vector_size(16)));
typedef __attribute__((address_space(1))) v4i* as1_v4i;   // global <4 x i32>*
typedef __attribute__((address_space(3))) v4i* as3_v4i;   // LDS    <4 x i32>*

__device__ __forceinline__ void async_cp16(const _Float16* g, _Float16* l) {
    // 16B per lane, global -> LDS, tracked by ASYNCcnt
    __builtin_amdgcn_global_load_async_to_lds_b128((as1_v4i)g, (as3_v4i)l, 0, 0);
}
__device__ __forceinline__ void async_wait0() {
    __builtin_amdgcn_s_wait_asynccnt(0);
}
#endif

__device__ __forceinline__ v8f wmma_f16(v16h a, v16h b, v8f c) {
    // D = A(16x32 f16) * B(32x16 f16) + C(16x16 f32)
    return __builtin_amdgcn_wmma_f32_16x16x32_f16(
        /*neg_a=*/false, a, /*neg_b=*/false, b,
        /*c_mod=*/(short)0, c, /*reuse_a=*/false, /*reuse_b=*/false);
}

// ---------------------------------------------------------------------------
// Kernel 1: LayerNorm over E, emit f16 normalized activations
// ---------------------------------------------------------------------------
__global__ void __launch_bounds__(256) ln_kernel(
    const float* __restrict__ x, const float* __restrict__ g,
    const float* __restrict__ bta, _Float16* __restrict__ xn)
{
    const int row  = blockIdx.x;              // 0 .. B*L-1
    const int t    = threadIdx.x;             // 256 threads, 4 elems each
    const int lane = t & 31, wid = t >> 5;
    const float* xr = x + (size_t)row * cE;

    float4 v = ((const float4*)xr)[t];
    float s  = v.x + v.y + v.z + v.w;
    float ss = v.x * v.x + v.y * v.y + v.z * v.z + v.w * v.w;
    #pragma unroll
    for (int off = 16; off; off >>= 1) {
        s  += __shfl_xor(s,  off, 32);
        ss += __shfl_xor(ss, off, 32);
    }
    __shared__ float rs[8], rss[8];
    if (lane == 0) { rs[wid] = s; rss[wid] = ss; }
    __syncthreads();
    if (wid == 0) {
        float a = (lane < 8) ? rs[lane]  : 0.f;
        float c = (lane < 8) ? rss[lane] : 0.f;
        #pragma unroll
        for (int off = 4; off; off >>= 1) {
            a += __shfl_xor(a, off, 32);
            c += __shfl_xor(c, off, 32);
        }
        if (lane == 0) { rs[0] = a; rss[0] = c; }
    }
    __syncthreads();
    const float mean = rs[0] * (1.f / cE);
    const float var  = rss[0] * (1.f / cE) - mean * mean;
    const float inv  = rsqrtf(var + 1e-5f);

    float4 gg = ((const float4*)g)[t];
    float4 bb = ((const float4*)bta)[t];
    h4 o;
    o[0] = (_Float16)((v.x - mean) * inv * gg.x + bb.x);
    o[1] = (_Float16)((v.y - mean) * inv * gg.y + bb.y);
    o[2] = (_Float16)((v.z - mean) * inv * gg.z + bb.z);
    o[3] = (_Float16)((v.w - mean) * inv * gg.w + bb.w);
    *(h4*)(xn + (size_t)row * cE + t * 4) = o;
}

// ---------------------------------------------------------------------------
// Kernel 2: f32 -> f16 conversion (weights)
// ---------------------------------------------------------------------------
__global__ void __launch_bounds__(256) cvt_kernel(
    const float* __restrict__ src, _Float16* __restrict__ dst, int n)
{
    int i = (blockIdx.x * 256 + threadIdx.x) * 4;
    if (i < n) {
        float4 v = *(const float4*)(src + i);
        h4 o;
        o[0] = (_Float16)v.x; o[1] = (_Float16)v.y;
        o[2] = (_Float16)v.z; o[3] = (_Float16)v.w;
        *(h4*)(dst + i) = o;
    }
}

// ---------------------------------------------------------------------------
// Kernel 3: QKV projection GEMM.  C[n,e] = sum_k xn[n,k] * W[e,k] + bias[e]
//   z=0 -> Q (scaled by D^-0.5), layout [b,h,l,d]
//   z=1 -> K,                    layout [b,h,l,d]
//   z=2 -> V,                    layout [b,h,d,l]   (transposed for PV WMMA)
// Block: 256 threads (8 waves), tile 128(M) x 128(N), K-step 32,
// double-buffered LDS with async global->LDS when available.
// ---------------------------------------------------------------------------
__global__ void __launch_bounds__(256) qkv_gemm(
    const _Float16* __restrict__ xn, const _Float16* __restrict__ wqkv,
    const float* __restrict__ bq, const float* __restrict__ bk,
    const float* __restrict__ bv,
    _Float16* __restrict__ Q, _Float16* __restrict__ K,
    _Float16* __restrict__ Vt)
{
    const int m0 = blockIdx.x * 128;
    const int n0 = blockIdx.y * 128;
    const int z  = blockIdx.z;
    const _Float16* w    = wqkv + (size_t)z * cE * cE;
    const float*    bias = (z == 0) ? bq : ((z == 1) ? bk : bv);

    __shared__ __align__(32) _Float16 At[2][128][32];
    __shared__ __align__(32) _Float16 Bt[2][128][32];

    const int t = threadIdx.x, lane = t & 31, wid = t >> 5;
    const int wm = wid & 3, wn = wid >> 2;       // 4 x 2 wave grid
    const int fm = lane & 15, fk = lane >> 4;

    const int row = t >> 1, cc = (t & 1) * 16;   // per-thread 32B stage chunk
    const _Float16* ga = xn + (size_t)(m0 + row) * cE + cc;
    const _Float16* gb = w  + (size_t)(n0 + row) * cE + cc;

    v8f acc[2][4];
    #pragma unroll
    for (int i = 0; i < 2; i++)
        #pragma unroll
        for (int j = 0; j < 4; j++) acc[i][j] = (v8f){};

#if USE_ASYNC
    // prologue: tile 0 in flight
    async_cp16(ga,     &At[0][row][cc]);
    async_cp16(ga + 8, &At[0][row][cc + 8]);
    async_cp16(gb,     &Bt[0][row][cc]);
    async_cp16(gb + 8, &Bt[0][row][cc + 8]);
#endif

    for (int ks = 0; ks < cE / 32; ks++) {
        const int cur = ks & 1;
#if USE_ASYNC
        const int nxt = cur ^ 1;
        async_wait0();            // my tile-`cur` DMAs done
        __syncthreads();          // everyone's done; prior reads of `nxt` done
        if (ks + 1 < cE / 32) {   // overlap next tile with this tile's WMMAs
            const _Float16* ga2 = ga + (ks + 1) * 32;
            const _Float16* gb2 = gb + (ks + 1) * 32;
            async_cp16(ga2,     &At[nxt][row][cc]);
            async_cp16(ga2 + 8, &At[nxt][row][cc + 8]);
            async_cp16(gb2,     &Bt[nxt][row][cc]);
            async_cp16(gb2 + 8, &Bt[nxt][row][cc + 8]);
        }
#else
        {
            const U128* sa = (const U128*)(ga + ks * 32);
            U128* da = (U128*)&At[cur][row][cc];
            da[0] = sa[0]; da[1] = sa[1];
            const U128* sb = (const U128*)(gb + ks * 32);
            U128* db = (U128*)&Bt[cur][row][cc];
            db[0] = sb[0]; db[1] = sb[1];
        }
        __syncthreads();
#endif
        v16h a[2], b[4];
        #pragma unroll
        for (int mi = 0; mi < 2; mi++) {
            const _Float16* p = &At[cur][wm * 32 + mi * 16 + fm][0];
            F16x16 u;
            u.u[0] = *(const U128*)(p + fk * 16);
            u.u[1] = *(const U128*)(p + 8 + fk * 16);
            a[mi] = u.v;
        }
        #pragma unroll
        for (int ni = 0; ni < 4; ni++)
            b[ni] = *(const v16h*)&Bt[cur][wn * 64 + ni * 16 + fm][fk * 16];

        #pragma unroll
        for (int mi = 0; mi < 2; mi++)
            #pragma unroll
            for (int ni = 0; ni < 4; ni++)
                acc[mi][ni] = wmma_f16(a[mi], b[ni], acc[mi][ni]);
    }

    const float scale = (z == 0) ? 0.125f : 1.0f;   // D^-0.5 folded into Q
    #pragma unroll
    for (int mi = 0; mi < 2; mi++) {
        #pragma unroll
        for (int ni = 0; ni < 4; ni++) {
            const int col = n0 + wn * 64 + ni * 16 + fm;
            const float bcol = bias[col];
            const int hh = col >> 6, dd = col & (cD - 1);
            #pragma unroll
            for (int r = 0; r < 8; r++) {
                const int m    = r + 8 * fk;
                const int grow = m0 + wm * 32 + mi * 16 + m;
                const int bb   = grow >> 11;          // / L
                const int ll   = grow & (cL - 1);
                const _Float16 hv = (_Float16)((acc[mi][ni][r] + bcol) * scale);
                if (z == 2)
                    Vt[(((size_t)bb * cH + hh) * cD + dd) * cL + ll] = hv;
                else if (z == 0)
                    Q[(((size_t)bb * cH + hh) * cL + ll) * cD + dd] = hv;
                else
                    K[(((size_t)bb * cH + hh) * cL + ll) * cD + dd] = hv;
            }
        }
    }
}

// ---------------------------------------------------------------------------
// Kernel 4: attention for one (b, h, 16-row q-tile).
//   Uses CDNA5's big LDS: full 16 x 2048 logits tile resident in LDS (f32).
// ---------------------------------------------------------------------------
__global__ void __launch_bounds__(256) attn_kernel(
    const _Float16* __restrict__ Q, const _Float16* __restrict__ K,
    const _Float16* __restrict__ Vt, const float* __restrict__ bias,
    const int* __restrict__ mask, const float* __restrict__ gate,
    _Float16* __restrict__ Y)
{
    const int q0 = blockIdx.x * 16;
    const int h  = blockIdx.y;
    const int b  = blockIdx.z;

    __shared__ __align__(32) float S[16][2064];      // logits / probs (padded)
    __shared__ float Ob[2][16][cD];                  // partial O (2 key halves)
    __shared__ float rowinv[16];

    const int t = threadIdx.x, lane = t & 31, wid = t >> 5;
    const int fm = lane & 15, fk = lane >> 4;
    const size_t bh = (size_t)b * cH + h;
    const _Float16* Qp = Q  + (bh * cL + q0) * cD;
    const _Float16* Kp = K  + bh * cL * cD;
    const _Float16* Vp = Vt + bh * cD * cL;

    // Get the 128 KB bias tile (the roofline-dominant stream) in flight now,
    // so it overlaps the phase-1 WMMAs.  16 rows x 8 KB = 1024 x 128B lines.
    {
        const float* bbase = bias + (bh * cL + (size_t)q0) * cL;
        for (int i = t; i < 1024; i += 256)
            __builtin_prefetch(bbase + (size_t)(i >> 6) * cL + (i & 63) * 32, 0, 1);
    }

    // ---- Phase 1: S = Qs @ K^T  (keys split 8 ways over waves) ----
    v16h aq[2];
    #pragma unroll
    for (int s2 = 0; s2 < 2; s2++) {
        const _Float16* p = Qp + fm * cD + s2 * 32;
        F16x16 u;
        u.u[0] = *(const U128*)(p + fk * 16);
        u.u[1] = *(const U128*)(p + 8 + fk * 16);
        aq[s2] = u.v;
    }
    const int keyw = wid * 256;
    for (int kt = 0; kt < 16; kt++) {
        const int key0 = keyw + kt * 16;
        v8f acc = (v8f){};
        #pragma unroll
        for (int s2 = 0; s2 < 2; s2++) {
            const _Float16* p = Kp + (size_t)(key0 + fm) * cD + s2 * 32 + fk * 16;
            acc = wmma_f16(aq[s2], *(const v16h*)p, acc);
        }
        #pragma unroll
        for (int r = 0; r < 8; r++)
            S[r + 8 * fk][key0 + fm] = acc[r];
    }
    __syncthreads();

    // ---- Phase 2: add gate*bias + pad, exact softmax (16 threads / row) ----
    const int tpr = t & 15, row = t >> 4;
    const float g = gate[h];
    const float* bp = bias + (bh * cL + (size_t)(q0 + row)) * cL;
    const int*   mp = mask + b * cL;
    float mx = -3.0e38f;
    for (int i = 0; i < cL / 16; i++) {
        const int k = tpr + (i << 4);
        float v = S[row][k] + g * bp[k] + (mp[k] ? 0.f : -1e9f);
        S[row][k] = v;
        mx = fmaxf(mx, v);
    }
    #pragma unroll
    for (int off = 8; off; off >>= 1) mx = fmaxf(mx, __shfl_xor(mx, off, 16));
    float sum = 0.f;
    for (int i = 0; i < cL / 16; i++) {
        const int k = tpr + (i << 4);
        float e = __expf(S[row][k] - mx);
        S[row][k] = e;
        sum += e;
    }
    #pragma unroll
    for (int off = 8; off; off >>= 1) sum += __shfl_xor(sum, off, 16);
    if (tpr == 0) rowinv[row] = 1.f / sum;
    __syncthreads();

    // ---- Phase 3: O = P @ V  (waves: 4 d-tiles x 2 key halves) ----
    const int ni = wid & 3, kh2 = wid >> 2;
    v8f oacc = (v8f){};
    for (int ks = 0; ks < 32; ks++) {
        const int kb = kh2 * 1024 + ks * 32;
        v16h pa;                                   // f32 LDS -> f16 A-frag
        const float* sp = &S[fm][kb + fk * 16];
        #pragma unroll
        for (int j = 0; j < 8; j++) {
            pa[j]     = (_Float16)sp[j];
            pa[8 + j] = (_Float16)sp[8 + j];
        }
        const _Float16* vp = Vp + (size_t)(ni * 16 + fm) * cL + kb + fk * 16;
        oacc = wmma_f16(pa, *(const v16h*)vp, oacc);
    }
    #pragma unroll
    for (int r = 0; r < 8; r++)
        Ob[kh2][r + 8 * fk][ni * 16 + fm] = oacc[r];
    __syncthreads();

    // ---- Epilogue: deterministic 2-way reduce, normalize, write f16 ----
    {
        const int idx = t * 4;                    // 16*64 = 1024 outputs
        const int m = idx >> 6, d0 = idx & (cD - 1);
        const float inv = rowinv[m];
        _Float16* yp = Y + ((size_t)b * cL + q0 + m) * cE + h * cD + d0;
        #pragma unroll
        for (int j = 0; j < 4; j++)
            yp[j] = (_Float16)((Ob[0][m][d0 + j] + Ob[1][m][d0 + j]) * inv);
    }
}

// ---------------------------------------------------------------------------
// Kernel 5: output projection + residual: out = x + Y @ wo^T + bo  (f32 out)
// ---------------------------------------------------------------------------
__global__ void __launch_bounds__(256) oproj_gemm(
    const _Float16* __restrict__ Y, const _Float16* __restrict__ wo,
    const float* __restrict__ bo, const float* __restrict__ x,
    float* __restrict__ out)
{
    const int m0 = blockIdx.x * 128;
    const int n0 = blockIdx.y * 128;

    __shared__ __align__(32) _Float16 At[2][128][32];
    __shared__ __align__(32) _Float16 Bt[2][128][32];

    const int t = threadIdx.x, lane = t & 31, wid = t >> 5;
    const int wm = wid & 3, wn = wid >> 2;
    const int fm = lane & 15, fk = lane >> 4;

    const int row = t >> 1, cc = (t & 1) * 16;
    const _Float16* ga = Y  + (size_t)(m0 + row) * cE + cc;
    const _Float16* gb = wo + (size_t)(n0 + row) * cE + cc;

    v8f acc[2][4];
    #pragma unroll
    for (int i = 0; i < 2; i++)
        #pragma unroll
        for (int j = 0; j < 4; j++) acc[i][j] = (v8f){};

#if USE_ASYNC
    async_cp16(ga,     &At[0][row][cc]);
    async_cp16(ga + 8, &At[0][row][cc + 8]);
    async_cp16(gb,     &Bt[0][row][cc]);
    async_cp16(gb + 8, &Bt[0][row][cc + 8]);
#endif

    for (int ks = 0; ks < cE / 32; ks++) {
        const int cur = ks & 1;
#if USE_ASYNC
        const int nxt = cur ^ 1;
        async_wait0();
        __syncthreads();
        if (ks + 1 < cE / 32) {
            const _Float16* ga2 = ga + (ks + 1) * 32;
            const _Float16* gb2 = gb + (ks + 1) * 32;
            async_cp16(ga2,     &At[nxt][row][cc]);
            async_cp16(ga2 + 8, &At[nxt][row][cc + 8]);
            async_cp16(gb2,     &Bt[nxt][row][cc]);
            async_cp16(gb2 + 8, &Bt[nxt][row][cc + 8]);
        }
#else
        {
            const U128* sa = (const U128*)(ga + ks * 32);
            U128* da = (U128*)&At[cur][row][cc];
            da[0] = sa[0]; da[1] = sa[1];
            const U128* sb = (const U128*)(gb + ks * 32);
            U128* db = (U128*)&Bt[cur][row][cc];
            db[0] = sb[0]; db[1] = sb[1];
        }
        __syncthreads();
#endif
        v16h a[2], b[4];
        #pragma unroll
        for (int mi = 0; mi < 2; mi++) {
            const _Float16* p = &At[cur][wm * 32 + mi * 16 + fm][0];
            F16x16 u;
            u.u[0] = *(const U128*)(p + fk * 16);
            u.u[1] = *(const U128*)(p + 8 + fk * 16);
            a[mi] = u.v;
        }
        #pragma unroll
        for (int ni = 0; ni < 4; ni++)
            b[ni] = *(const v16h*)&Bt[cur][wn * 64 + ni * 16 + fm][fk * 16];

        #pragma unroll
        for (int mi = 0; mi < 2; mi++)
            #pragma unroll
            for (int ni = 0; ni < 4; ni++)
                acc[mi][ni] = wmma_f16(a[mi], b[ni], acc[mi][ni]);
    }

    #pragma unroll
    for (int mi = 0; mi < 2; mi++) {
        #pragma unroll
        for (int ni = 0; ni < 4; ni++) {
            const int col = n0 + wn * 64 + ni * 16 + fm;
            const float bcol = bo[col];
            #pragma unroll
            for (int r = 0; r < 8; r++) {
                const int grow = m0 + wm * 32 + mi * 16 + r + 8 * fk;
                const size_t idx = (size_t)grow * cE + col;
                out[idx] = acc[mi][ni][r] + bcol + x[idx];
            }
        }
    }
}

// ---------------------------------------------------------------------------
// Host launcher
// ---------------------------------------------------------------------------
extern "C" void kernel_launch(void* const* d_in, const int* in_sizes, int n_in,
                              void* d_out, int out_size, void* d_ws, size_t ws_size,
                              hipStream_t stream)
{
    const float* x    = (const float*)d_in[0];
    const float* bias = (const float*)d_in[1];
    const int*   mask = (const int*)d_in[2];
    const float* wq   = (const float*)d_in[3];
    const float* bq   = (const float*)d_in[4];
    const float* wk   = (const float*)d_in[5];
    const float* bk   = (const float*)d_in[6];
    const float* wv   = (const float*)d_in[7];
    const float* bv   = (const float*)d_in[8];
    const float* wo   = (const float*)d_in[9];
    const float* bo   = (const float*)d_in[10];
    const float* gate = (const float*)d_in[11];
    const float* ln_g = (const float*)d_in[12];
    const float* ln_b = (const float*)d_in[13];
    float* out = (float*)d_out;

    char* ws = (char*)d_ws;
    _Float16* xn16   = (_Float16*)(ws);                          //  8 MB
    _Float16* wqkv16 = (_Float16*)(ws + ((size_t)8  << 20));     //  6 MB
    _Float16* wo16   = (_Float16*)(ws + ((size_t)14 << 20));     //  2 MB
    _Float16* Q16    = (_Float16*)(ws + ((size_t)16 << 20));     //  8 MB
    _Float16* K16    = (_Float16*)(ws + ((size_t)24 << 20));     //  8 MB
    _Float16* Vt16   = (_Float16*)(ws + ((size_t)32 << 20));     //  8 MB
    _Float16* Y16    = (_Float16*)(ws + ((size_t)40 << 20));     //  8 MB (48 total)

    ln_kernel<<<cB * cL, 256, 0, stream>>>(x, ln_g, ln_b, xn16);
    cvt_kernel<<<(cE * cE) / 1024, 256, 0, stream>>>(wq, wqkv16,               cE * cE);
    cvt_kernel<<<(cE * cE) / 1024, 256, 0, stream>>>(wk, wqkv16 + cE * cE,     cE * cE);
    cvt_kernel<<<(cE * cE) / 1024, 256, 0, stream>>>(wv, wqkv16 + 2 * cE * cE, cE * cE);
    cvt_kernel<<<(cE * cE) / 1024, 256, 0, stream>>>(wo, wo16,                 cE * cE);

    qkv_gemm<<<dim3((cB * cL) / 128, cE / 128, 3), 256, 0, stream>>>(
        xn16, wqkv16, bq, bk, bv, Q16, K16, Vt16);

    attn_kernel<<<dim3(cL / 16, cH, cB), 256, 0, stream>>>(
        Q16, K16, Vt16, bias, mask, gate, Y16);

    oproj_gemm<<<dim3((cB * cL) / 128, cE / 128), 256, 0, stream>>>(
        Y16, wo16, bo, x, out);
}
